// FrameAttentionFusion_24550033063999
// MI455X (gfx1250) — matrix operand order repaired and checked by reading, not compiled, not run on hardware
//
#include <hip/hip_runtime.h>
#include <hip/hip_bf16.h>

typedef __bf16 bf16_t;
typedef __attribute__((ext_vector_type(8)))  __bf16 v8bf;
typedef __attribute__((ext_vector_type(16))) __bf16 v16bf;
typedef __attribute__((ext_vector_type(8)))  float  v8f;

#define C_DIM 512
#define HW 9216            // 96*96
#define LDA 520            // bf16 row stride for 64x512 buffers (pad -> conflict-free)
#define LDV 72             // token stride for transposed V / attn scratch
#define LDS_X 0            // t (bf16), later x1 (bf16)         64*520*2 = 66560
#define LDS_Q 66560        // Q (bf16), later attn-out, later h  64*520*2 = 66560
#define LDS_K 133120       // K (bf16)                           64*520*2 = 66560
#define LDS_V 199680       // V^T (bf16) [512][72]              512*72*2 = 73728
#define LDS_A 273408       // per-wave attn scratch [8][16][72]  8*16*72*2 = 18432
#define SMEM_BYTES 291840
// fp32 scratch [64][512] = 131072 bytes overlays K+V (dead after attention)

#define WMMA_BF16(a, b, c) \
  __builtin_amdgcn_wmma_f32_16x16x32_bf16(false, (a), false, (b), (short)0, (c), false, false)

static __device__ __forceinline__ v8f vzero8() {
  v8f v = {0.f,0.f,0.f,0.f,0.f,0.f,0.f,0.f};
  return v;
}

// A-matrix 16x32 bf16 fragment from LDS (row-major [rows][lda]).
// lanes 0-15: M=lane, K=[k0..k0+16); lanes 16-31: same M, K=[k0+16..k0+32).
static __device__ __forceinline__ v16bf frag_a_lds(const bf16_t* base, int lda, int mt, int k0) {
  const int lane = threadIdx.x & 31;
  const bf16_t* p = base + (mt*16 + (lane & 15))*lda + k0 + ((lane >> 4) << 4);
  v8bf lo = *(const v8bf*)p;
  v8bf hi = *(const v8bf*)(p + 8);
  return __builtin_shufflevector(lo, hi, 0,1,2,3,4,5,6,7,8,9,10,11,12,13,14,15);
}

// B-matrix 32x16 bf16 fragment from LDS: lane%16 = N column, 16 contiguous K per lane-half.
static __device__ __forceinline__ v16bf frag_b_lds(const bf16_t* base, int ldb, int n0, int k0) {
  const int lane = threadIdx.x & 31;
  const bf16_t* p = base + (n0 + (lane & 15))*ldb + k0 + ((lane >> 4) << 4);
  v8bf lo = *(const v8bf*)p;
  v8bf hi = *(const v8bf*)(p + 8);
  return __builtin_shufflevector(lo, hi, 0,1,2,3,4,5,6,7,8,9,10,11,12,13,14,15);
}

// B fragment from global W^T (row-major [N][K], 32B aligned) -> one 32-byte load per lane.
static __device__ __forceinline__ v16bf frag_b_glb(const bf16_t* base, int ldb, int n0, int k0) {
  const int lane = threadIdx.x & 31;
  const bf16_t* p = base + (n0 + (lane & 15))*ldb + k0 + ((lane >> 4) << 4);
  return *(const v16bf*)p;
}

// 64x64 output block per wave over K=512: per k-step load 4 A + 4 B fragments,
// then issue 16 independent WMMAs (each fragment amortized over 4 WMMAs).
static __device__ __forceinline__ void gemm_col64(const bf16_t* A, int lda,
                                                  const bf16_t* Bt, int ldb, int n0,
                                                  v8f acc[4][4]) {
  #pragma unroll
  for (int mt = 0; mt < 4; ++mt)
    #pragma unroll
    for (int nt = 0; nt < 4; ++nt) acc[mt][nt] = vzero8();
  #pragma unroll 2
  for (int kt = 0; kt < 16; ++kt) {
    v16bf a[4], b[4];
    #pragma unroll
    for (int mt = 0; mt < 4; ++mt) a[mt] = frag_a_lds(A, lda, mt, kt*32);
    #pragma unroll
    for (int nt = 0; nt < 4; ++nt) b[nt] = frag_b_glb(Bt, ldb, n0 + nt*16, kt*32);
    #pragma unroll
    for (int mt = 0; mt < 4; ++mt)
      #pragma unroll
      for (int nt = 0; nt < 4; ++nt)
        acc[mt][nt] = WMMA_BF16(a[mt], b[nt], acc[mt][nt]);
  }
}

// Per-row LayerNorm on fp32 [64][512]; one row per wave iteration, shfl reductions.
static __device__ __forceinline__ void ln_rows(float* Fp, bf16_t* Xout,
                                               const float* g, const float* be) {
  const int wave = threadIdx.x >> 5, lane = threadIdx.x & 31;
  for (int row = wave; row < 64; row += 8) {
    float s = 0.f, s2 = 0.f;
    for (int n = lane; n < 512; n += 32) {
      float v = Fp[row*512 + n];
      s += v; s2 += v*v;
    }
    #pragma unroll
    for (int off = 1; off < 32; off <<= 1) {
      s  += __shfl_xor(s,  off, 32);
      s2 += __shfl_xor(s2, off, 32);
    }
    float mean = s * (1.f/512.f);
    float var  = fmaxf(s2 * (1.f/512.f) - mean*mean, 0.f);
    float rstd = rsqrtf(var + 1e-5f);
    for (int n = lane; n < 512; n += 32) {
      float v = (Fp[row*512 + n] - mean) * rstd * g[n] + be[n];
      Fp[row*512 + n] = v;
      if (Xout) Xout[row*LDA + n] = (bf16_t)v;
    }
  }
}

__global__ void cvt_transpose(const float* __restrict__ in, bf16_t* __restrict__ out,
                              int K, int N) {
  int idx = blockIdx.x * 256 + threadIdx.x;
  if (idx >= K * N) return;
  int n = idx / K, k = idx % K;          // out[n][k] = in[k][n], coalesced writes
  out[idx] = (bf16_t)in[k * N + n];
}

__global__ __launch_bounds__(256, 1)
void fattn_block(const float* __restrict__ x,
                 const float* __restrict__ bqkv, const float* __restrict__ bo,
                 const float* __restrict__ g1, const float* __restrict__ be1,
                 const float* __restrict__ bf1v, const float* __restrict__ bf2v,
                 const float* __restrict__ g2, const float* __restrict__ be2,
                 const bf16_t* __restrict__ WqkvT, const bf16_t* __restrict__ WoT,
                 const bf16_t* __restrict__ W1T, const bf16_t* __restrict__ W2T,
                 float* __restrict__ out)
{
  extern __shared__ char smem[];
  bf16_t* Xb = (bf16_t*)(smem + LDS_X);
  bf16_t* Qb = (bf16_t*)(smem + LDS_Q);
  bf16_t* Kb = (bf16_t*)(smem + LDS_K);
  bf16_t* Vt = (bf16_t*)(smem + LDS_V);
  bf16_t* As = (bf16_t*)(smem + LDS_A);
  float*  Fp = (float*)(smem + LDS_K);   // overlays K+V after attention

  const int tid  = threadIdx.x;
  const int wave = tid >> 5;
  const int lane = tid & 31;
  const int half = lane >> 4;
  const int nl   = lane & 15;
  const int col0 = wave * 64;            // this wave's 64-column slab

  const int wid  = blockIdx.x;
  const int bimg = wid / 144;
  const int wrc  = wid % 144;
  const size_t base = (size_t)bimg * C_DIM * HW
                    + (size_t)(wrc / 12) * 8 * 96 + (size_t)(wrc % 12) * 8;

  // ---- Phase 0: load window tokens, convert to bf16 ----
  for (int i = tid; i < 64*C_DIM; i += 256) {
    int m = i & 63, ch = i >> 6;
    Xb[m*LDA + ch] = (bf16_t)x[base + (size_t)ch*HW + (m >> 3)*96 + (m & 7)];
  }
  __syncthreads();

  // ---- GEMM1: qkv = t @ Wqkv + bqkv.  Wave w computes head w's Q, K, V slabs. ----
  {
    v8f acc[4][4];
    // Q columns [col0, col0+64)
    gemm_col64(Xb, LDA, WqkvT, C_DIM, col0, acc);
    #pragma unroll
    for (int nt = 0; nt < 4; ++nt) {
      int n = col0 + nt*16 + nl;
      float bias = bqkv[n];
      #pragma unroll
      for (int mt = 0; mt < 4; ++mt)
        #pragma unroll
        for (int r = 0; r < 8; ++r)
          Qb[(mt*16 + half*8 + r)*LDA + n] = (bf16_t)(acc[mt][nt][r] + bias);
    }
    // K columns [512+col0, 512+col0+64)
    gemm_col64(Xb, LDA, WqkvT, C_DIM, 512 + col0, acc);
    #pragma unroll
    for (int nt = 0; nt < 4; ++nt) {
      int n = col0 + nt*16 + nl;
      float bias = bqkv[512 + n];
      #pragma unroll
      for (int mt = 0; mt < 4; ++mt)
        #pragma unroll
        for (int r = 0; r < 8; ++r)
          Kb[(mt*16 + half*8 + r)*LDA + n] = (bf16_t)(acc[mt][nt][r] + bias);
    }
    // V columns [1024+col0, 1024+col0+64) -> stored transposed
    gemm_col64(Xb, LDA, WqkvT, C_DIM, 1024 + col0, acc);
    #pragma unroll
    for (int nt = 0; nt < 4; ++nt) {
      int n = col0 + nt*16 + nl;
      float bias = bqkv[1024 + n];
      #pragma unroll
      for (int mt = 0; mt < 4; ++mt)
        #pragma unroll
        for (int r = 0; r < 8; ++r)
          Vt[n*LDV + (mt*16 + half*8 + r)] = (bf16_t)(acc[mt][nt][r] + bias);
    }
  }
  __syncthreads();

  // ---- Attention: one head per wave (all inputs are wave-local slabs) ----
  {
    const bf16_t* Qh = Qb + col0;
    const bf16_t* Kh = Kb + col0;
    const bf16_t* Vh = Vt + col0*LDV;
    bf16_t* Aw = As + wave*16*LDV;
    for (int mt = 0; mt < 4; ++mt) {
      v8f s0 = vzero8(), s1 = vzero8(), s2 = vzero8(), s3 = vzero8();
      #pragma unroll
      for (int kt = 0; kt < 2; ++kt) {
        v16bf a = frag_a_lds(Qh, LDA, mt, kt*32);
        s0 = WMMA_BF16(a, frag_b_lds(Kh, LDA, 0,  kt*32), s0);  // B = K^T (contiguous d)
        s1 = WMMA_BF16(a, frag_b_lds(Kh, LDA, 16, kt*32), s1);
        s2 = WMMA_BF16(a, frag_b_lds(Kh, LDA, 32, kt*32), s2);
        s3 = WMMA_BF16(a, frag_b_lds(Kh, LDA, 48, kt*32), s3);
      }
      // softmax over the 64 columns of this 16-row block (rows live across 16-lane halves)
      #pragma unroll
      for (int r = 0; r < 8; ++r) {
        float v0 = s0[r]*0.125f, v1 = s1[r]*0.125f, v2 = s2[r]*0.125f, v3 = s3[r]*0.125f;
        float mx = fmaxf(fmaxf(v0,v1), fmaxf(v2,v3));
        #pragma unroll
        for (int off = 1; off < 16; off <<= 1) mx = fmaxf(mx, __shfl_xor(mx, off, 32));
        float e0 = __expf(v0-mx), e1 = __expf(v1-mx), e2 = __expf(v2-mx), e3 = __expf(v3-mx);
        float sm = e0 + e1 + e2 + e3;
        #pragma unroll
        for (int off = 1; off < 16; off <<= 1) sm += __shfl_xor(sm, off, 32);
        float inv = 1.0f / sm;
        int rl = half*8 + r;
        Aw[rl*LDV +      nl] = (bf16_t)(e0*inv);
        Aw[rl*LDV + 16 + nl] = (bf16_t)(e1*inv);
        Aw[rl*LDV + 32 + nl] = (bf16_t)(e2*inv);
        Aw[rl*LDV + 48 + nl] = (bf16_t)(e3*inv);
      }
      // o[mt] = attn @ V  (A from per-wave scratch; B = V^T rows -> contiguous tokens)
      v8f o0 = vzero8(), o1 = vzero8(), o2 = vzero8(), o3 = vzero8();
      #pragma unroll
      for (int kt = 0; kt < 2; ++kt) {
        v16bf a = frag_a_lds(Aw, LDV, 0, kt*32);
        o0 = WMMA_BF16(a, frag_b_lds(Vh, LDV, 0,  kt*32), o0);
        o1 = WMMA_BF16(a, frag_b_lds(Vh, LDV, 16, kt*32), o1);
        o2 = WMMA_BF16(a, frag_b_lds(Vh, LDV, 32, kt*32), o2);
        o3 = WMMA_BF16(a, frag_b_lds(Vh, LDV, 48, kt*32), o3);
      }
      #pragma unroll
      for (int r = 0; r < 8; ++r) {
        int m = mt*16 + half*8 + r;                 // rows mt already consumed for scores
        bf16_t* dst = Qb + m*LDA + col0;            // reuse Q slice as attn-out
        dst[nl]      = (bf16_t)o0[r];
        dst[16 + nl] = (bf16_t)o1[r];
        dst[32 + nl] = (bf16_t)o2[r];
        dst[48 + nl] = (bf16_t)o3[r];
      }
    }
  }
  __syncthreads();

  // ---- GEMM2: o @ Wo + bo + t -> fp32 scratch ----
  {
    v8f acc[4][4];
    gemm_col64(Qb, LDA, WoT, C_DIM, col0, acc);
    #pragma unroll
    for (int nt = 0; nt < 4; ++nt) {
      int n = col0 + nt*16 + nl;
      float bias = bo[n];
      #pragma unroll
      for (int mt = 0; mt < 4; ++mt)
        #pragma unroll
        for (int r = 0; r < 8; ++r) {
          int m = mt*16 + half*8 + r;
          Fp[m*512 + n] = acc[mt][nt][r] + bias + (float)Xb[m*LDA + n];
        }
    }
  }
  __syncthreads();

  // ---- LN1: x1 fp32 in Fp, x1 bf16 into Xb ----
  ln_rows(Fp, Xb, g1, be1);
  __syncthreads();

  // ---- GEMM3: h = relu(x1 @ W1 + bf1) -> bf16 into Qb ----
  {
    v8f acc[4][4];
    gemm_col64(Xb, LDA, W1T, C_DIM, col0, acc);
    #pragma unroll
    for (int nt = 0; nt < 4; ++nt) {
      int n = col0 + nt*16 + nl;
      float bias = bf1v[n];
      #pragma unroll
      for (int mt = 0; mt < 4; ++mt)
        #pragma unroll
        for (int r = 0; r < 8; ++r)
          Qb[(mt*16 + half*8 + r)*LDA + n] = (bf16_t)fmaxf(acc[mt][nt][r] + bias, 0.f);
    }
  }
  __syncthreads();

  // ---- GEMM4: f = h @ W2 + bf2 + x1 -> fp32 in place ----
  {
    v8f acc[4][4];
    gemm_col64(Qb, LDA, W2T, C_DIM, col0, acc);
    #pragma unroll
    for (int nt = 0; nt < 4; ++nt) {
      int n = col0 + nt*16 + nl;
      float bias = bf2v[n];
      #pragma unroll
      for (int mt = 0; mt < 4; ++mt)
        #pragma unroll
        for (int r = 0; r < 8; ++r) {
          int m = mt*16 + half*8 + r;
          Fp[m*512 + n] = acc[mt][nt][r] + bias + Fp[m*512 + n];
        }
    }
  }
  __syncthreads();

  // ---- LN2 ----
  ln_rows(Fp, nullptr, g2, be2);
  __syncthreads();

  // ---- Writeout ----
  for (int i = tid; i < 64*C_DIM; i += 256) {
    int m = i & 63, ch = i >> 6;
    out[base + (size_t)ch*HW + (m >> 3)*96 + (m & 7)] = Fp[m*512 + ch];
  }
}

extern "C" void kernel_launch(void* const* d_in, const int* in_sizes, int n_in,
                              void* d_out, int out_size, void* d_ws, size_t ws_size,
                              hipStream_t stream) {
  const float* x    = (const float*)d_in[0];
  const float* Wqkv = (const float*)d_in[1];
  const float* bqkv = (const float*)d_in[2];
  const float* Wo   = (const float*)d_in[3];
  const float* bo   = (const float*)d_in[4];
  const float* g1   = (const float*)d_in[5];
  const float* be1  = (const float*)d_in[6];
  const float* W1   = (const float*)d_in[7];
  const float* bf1  = (const float*)d_in[8];
  const float* W2   = (const float*)d_in[9];
  const float* bf2  = (const float*)d_in[10];
  const float* g2   = (const float*)d_in[11];
  const float* be2  = (const float*)d_in[12];
  float* out = (float*)d_out;

  bf16_t* ws    = (bf16_t*)d_ws;
  bf16_t* WqkvT = ws;                       // [1536][512]
  bf16_t* WoT   = WqkvT + 512 * 1536;       // [512][512]
  bf16_t* W1T   = WoT + 512 * 512;
  bf16_t* W2T   = W1T + 512 * 512;

  cvt_transpose<<<(512*1536 + 255) / 256, 256, 0, stream>>>(Wqkv, WqkvT, 512, 1536);
  cvt_transpose<<<(512*512  + 255) / 256, 256, 0, stream>>>(Wo,   WoT,   512, 512);
  cvt_transpose<<<(512*512  + 255) / 256, 256, 0, stream>>>(W1,   W1T,   512, 512);
  cvt_transpose<<<(512*512  + 255) / 256, 256, 0, stream>>>(W2,   W2T,   512, 512);

  int batches = in_sizes[0] / (512 * 96 * 96);
  fattn_block<<<dim3(batches * 144), 256, SMEM_BYTES, stream>>>(
      x, bqkv, bo, g1, be1, bf1, bf2, g2, be2, WqkvT, WoT, W1T, W2T, out);
}
